// VectorQuantizer_62302795595989
// MI455X (gfx1250) — matrix operand order repaired and checked by reading, not compiled, not run on hardware
//
#include <hip/hip_runtime.h>
#include <hip/hip_bf16.h>

typedef __bf16 v16bf __attribute__((ext_vector_type(16)));
typedef __bf16 v8bf  __attribute__((ext_vector_type(8)));
typedef float  v8f   __attribute__((ext_vector_type(8)));
typedef unsigned int u32x4 __attribute__((ext_vector_type(4)));
typedef int          i32x4 __attribute__((ext_vector_type(4)));
typedef int          i32x8 __attribute__((ext_vector_type(8)));

#define K_EMB    1024
#define D_DIM    256
#define T_DIM    4096
#define B_DIM    16
#define N_ROWS   (B_DIM * T_DIM)                      // 65536
#define TILE_PAD 264                                  // 256 + 8 halfs: 16B-aligned runs, broken bank conflicts
#define WAVES_PER_BLOCK 4
#define ROWS_PER_WAVE   32                            // two 16-row M-tiles per wave
#define MAIN_BLOCKS (N_ROWS / (ROWS_PER_WAVE * WAVES_PER_BLOCK)) // 512
#define KT_TILES  (K_EMB / 16)                        // 64
#define BTILE_SH  4096                                // shorts per 16-code B tile (8 KB)

#if defined(__has_builtin)
#if __has_builtin(__builtin_amdgcn_tensor_load_to_lds) && __has_builtin(__builtin_amdgcn_s_wait_tensorcnt)
#define HAVE_TDM 1
#endif
#endif

__device__ __forceinline__ unsigned short f32_bf16(float f) {
    unsigned int u = __float_as_uint(f);
    u += 0x7FFFu + ((u >> 16) & 1u);                  // round-to-nearest-even
    return (unsigned short)(u >> 16);
}

#if defined(HAVE_TDM)
// 1-D TDM copy: 8192 bytes global -> LDS. D# per CDNA5 ISA ch.8:
//   group0: count=1 | lds_addr | global_addr[56:0] | type=2
//   group1: workgroup_mask=0, data_size=3 (8B), tensor_dim0=1024, tensor_dim1=1,
//           tile_dim0=1024, tile_dim1=0 (1-D), tensor_dim0_stride=1024
// This toolchain declares the 6-arg builtin: (g0, g1, g2, g3, g4, cpol); groups 2-4 unused here.
__device__ __forceinline__ void tdm_load_b_tile(unsigned lds_off, const unsigned short* gsrc) {
    const unsigned long long ga = (unsigned long long)(size_t)gsrc;
    u32x4 g0;
    g0[0] = 1u;
    g0[1] = lds_off;
    g0[2] = (unsigned)ga;
    g0[3] = (unsigned)((ga >> 32) & 0x01FFFFFFu) | (2u << 30);
    i32x8 g1;
    g1[0] = 3 << 16;                 // data_size = 8B
    g1[1] = (int)(1024u << 16);      // tensor_dim0[15:0]
    g1[2] = 1 << 16;                 // tensor_dim0[31:16]=0 | tensor_dim1[15:0]=1
    g1[3] = (int)(1024u << 16);      // tensor_dim1[31:16]=0 | tile_dim0=1024
    g1[4] = 0;                       // tile_dim1=0 (1-D), tile_dim2=0
    g1[5] = 1024;                    // tensor_dim0_stride[31:0]
    g1[6] = 0;
    g1[7] = 0;
    const i32x4 gz4 = {0, 0, 0, 0};
    const i32x8 gz8 = {0, 0, 0, 0, 0, 0, 0, 0};
    __builtin_amdgcn_tensor_load_to_lds(g0, g1, gz4, gz4, gz8, 0);
}
#endif

// ||e||^2 per codebook row. One wave per row.
__global__ void vq_enorm(const float* __restrict__ emb, float* __restrict__ enorm) {
    const int k = blockIdx.x, l = threadIdx.x;
    const float* row = emb + (size_t)k * D_DIM;
    float s = 0.f;
    #pragma unroll
    for (int i = 0; i < D_DIM; i += 32) { float v = row[i + l]; s += v * v; }
    #pragma unroll
    for (int off = 16; off; off >>= 1) s += __shfl_xor(s, off, 32);
    if (l == 0) enorm[k] = s;
}

// Repack emb (f32 row-major) -> bf16 B-fragment layout:
// flat = ((kt*8 + s)*32 + lane)*16 + j ; column e = kt*16+(lane&15); K = s*32 + 16*(j>>3) + 8*((lane>>4)&1) + (j&7)
__global__ void vq_pack_b(const float* __restrict__ emb, unsigned short* __restrict__ embB) {
    const int flat = blockIdx.x * blockDim.x + threadIdx.x;    // 0..262143
    const int j    = flat & 15;
    const int lane = (flat >> 4) & 31;
    const int s    = (flat >> 9) & 7;
    const int kt   = flat >> 12;
    const int e    = kt * 16 + (lane & 15);
    const int kk   = s * 32 + (j >> 3) * 16 + ((lane >> 4) & 1) * 8 + (j & 7);
    embB[flat] = f32_bf16(emb[(size_t)e * D_DIM + kk]);
}

__global__ __launch_bounds__(128) void vq_main(
    const float* __restrict__ z, const float* __restrict__ emb,
    const unsigned short* __restrict__ embB, const float* __restrict__ enorm,
    float* __restrict__ out, float* __restrict__ partials)
{
    __shared__ unsigned short ztile[WAVES_PER_BLOCK][16 * TILE_PAD]; // 33792 B (reused for both M-tiles)
    __shared__ unsigned short bbuf[2][BTILE_SH];                     // 16384 B double-buffered B tiles
    __shared__ float s_enorm[K_EMB];                                 //  4096 B
    __shared__ int   s_idx[WAVES_PER_BLOCK][ROWS_PER_WAVE];
    __shared__ float s_wsum[WAVES_PER_BLOCK];

    const int tid  = threadIdx.x;
    const int w    = tid >> 5;
    const int l    = tid & 31;
    const int r    = l & 15;
    const int half = (l >> 4) & 1;

#if defined(HAVE_TDM)
    // Low 32 bits of a flat LDS pointer are the LDS byte offset (ISA flat-aperture rule).
    const unsigned lds_b0 = (unsigned)(size_t)&bbuf[0][0];
    const unsigned lds_b1 = (unsigned)(size_t)&bbuf[1][0];
    if (tid < 32) tdm_load_b_tile(lds_b0, embB);   // prefetch kt=0, overlaps phase 1
#endif

    #pragma unroll
    for (int i = 0; i < K_EMB; i += 128) s_enorm[i + tid] = enorm[i + tid];

    const int n0 = (blockIdx.x * WAVES_PER_BLOCK + w) * ROWS_PER_WAVE;
    const int b  = n0 >> 12;                         // n = b*T + t  (32 rows never cross b)
    const int t0 = n0 & (T_DIM - 1);
    const float* zb = z + (size_t)b * D_DIM * T_DIM;

    // ---- Phase 1: stage each 16-row tile in LDS (bf16, [r][d]), hoist A fragments.
    //      Same LDS buffer reused for tile 1: LDS ops are in-order per wave (no WAR hazard).
    v16bf af[2][8];
    {
        const int d_lane = l >> 2;
        const int rr     = (l & 3) * 4;
        const int koff   = half * 8;
        const unsigned short* abase = &ztile[w][r * TILE_PAD];
        #pragma unroll
        for (int tt = 0; tt < 2; ++tt) {
            const int t0s = t0 + tt * 16;
            for (int dd = 0; dd < D_DIM; dd += 8) {
                const int d = dd + d_lane;
                const float4 v = *(const float4*)(zb + (size_t)d * T_DIM + t0s + rr);
                ztile[w][(rr + 0) * TILE_PAD + d] = f32_bf16(v.x);
                ztile[w][(rr + 1) * TILE_PAD + d] = f32_bf16(v.y);
                ztile[w][(rr + 2) * TILE_PAD + d] = f32_bf16(v.z);
                ztile[w][(rr + 3) * TILE_PAD + d] = f32_bf16(v.w);
            }
            // A 16x32 layout: half 0 holds K-runs {s*32+0..7, +16..23}; half 1 {+8..15, +24..31}
            #pragma unroll
            for (int s = 0; s < 8; ++s) {
                v8bf a0 = *(const v8bf*)(abase + s * 32 + koff);
                v8bf a1 = *(const v8bf*)(abase + s * 32 + koff + 16);
                af[tt][s] = __builtin_shufflevector(a0, a1,
                    0,1,2,3,4,5,6,7,8,9,10,11,12,13,14,15);
            }
        }
    }
    __syncthreads();

    float minv0[8], minv1[8]; int mini0[8], mini1[8];
    #pragma unroll
    for (int v = 0; v < 8; ++v) {
        minv0[v] = __builtin_huge_valf(); mini0[v] = 0;
        minv1[v] = __builtin_huge_valf(); mini1[v] = 0;
    }

    // ---- Distance sweep: 64 K-tiles; B tile streamed into LDS once per BLOCK (shared by 4 waves) ----
    for (int kt = 0; kt < KT_TILES; ++kt) {
        const int pp = kt & 1;
#if defined(HAVE_TDM)
        if (tid < 32) __builtin_amdgcn_s_wait_tensorcnt(0);  // tile kt landed in bbuf[pp]
        __syncthreads();                                     // publish kt; bbuf[pp^1] free
        if (tid < 32 && kt + 1 < KT_TILES)
            tdm_load_b_tile(pp ? lds_b0 : lds_b1, embB + (size_t)(kt + 1) * BTILE_SH);
#else
        __syncthreads();                                     // previous tile fully consumed
        {   // cooperative synchronous copy: 128 threads x 64B
            const float4* src = (const float4*)(embB + (size_t)kt * BTILE_SH);
            float4* dst = (float4*)&bbuf[pp][0];
            #pragma unroll
            for (int i = 0; i < 4; ++i) dst[tid + i * 128] = src[tid + i * 128];
        }
        __syncthreads();
#endif
        v8f c0 = {0.f, 0.f, 0.f, 0.f, 0.f, 0.f, 0.f, 0.f};
        v8f c1 = {0.f, 0.f, 0.f, 0.f, 0.f, 0.f, 0.f, 0.f};
        const unsigned short* bb = &bbuf[pp][(size_t)l * 16];
        #pragma unroll
        for (int s = 0; s < 8; ++s) {
            v16bf bf = *(const v16bf*)(bb + s * 512);        // 2x ds_load_b128
            c0 = __builtin_amdgcn_wmma_f32_16x16x32_bf16(false, af[0][s], false, bf,
                                                         (short)0, c0, false, false);
            c1 = __builtin_amdgcn_wmma_f32_16x16x32_bf16(false, af[1][s], false, bf,
                                                         (short)0, c1, false, false);
        }
        // C layout: lane holds row M = v + 8*half, column e = kt*16 + (l&15)
        const int   e  = kt * 16 + r;
        const float en = s_enorm[e];
        #pragma unroll
        for (int v = 0; v < 8; ++v) {
            const float d0 = en - 2.0f * c0[v];
            if (d0 < minv0[v]) { minv0[v] = d0; mini0[v] = e; }  // e increasing -> first-min kept
            const float d1 = en - 2.0f * c1[v];
            if (d1 < minv1[v]) { minv1[v] = d1; mini1[v] = e; }
        }
    }

    // ---- Argmin across the 16 lanes sharing each row (tie -> lower index) ----
    #pragma unroll
    for (int off = 1; off <= 8; off <<= 1) {
        #pragma unroll
        for (int v = 0; v < 8; ++v) {
            float ov = __shfl_xor(minv0[v], off, 32);
            int   oi = __shfl_xor(mini0[v], off, 32);
            if (ov < minv0[v] || (ov == minv0[v] && oi < mini0[v])) { minv0[v] = ov; mini0[v] = oi; }
            ov = __shfl_xor(minv1[v], off, 32);
            oi = __shfl_xor(mini1[v], off, 32);
            if (ov < minv1[v] || (ov == minv1[v] && oi < mini1[v])) { minv1[v] = ov; mini1[v] = oi; }
        }
    }
    if (l == 0 || l == 16) {
        #pragma unroll
        for (int v = 0; v < 8; ++v) {
            s_idx[w][half * 8 + v]      = mini0[v];
            s_idx[w][16 + half * 8 + v] = mini1[v];
        }
    }
    __syncthreads();

    // ---- Phase 3: exact-f32 gather, [B,D,T] output, loss partial ----
    float acc = 0.f;
    #pragma unroll
    for (int tt = 0; tt < 2; ++tt) {
        const int eidx = s_idx[w][tt * 16 + r];
        const float* qrow = emb + (size_t)eidx * D_DIM;
        float* ob = out + (size_t)b * D_DIM * T_DIM;
        const int t0s = t0 + tt * 16;
        for (int dd = 0; dd < D_DIM; dd += 2) {
            const int d = dd + half;
            const float q = qrow[d];
            const size_t off = (size_t)d * T_DIM + t0s + r;   // lanes 0-15 coalesce along t
            const float zv = zb[off];
            ob[off] = q;
            const float df = q - zv;
            acc += df * df;
        }
    }
    #pragma unroll
    for (int off = 16; off; off >>= 1) acc += __shfl_xor(acc, off, 32);
    if (l == 0) s_wsum[w] = acc;
    __syncthreads();
    if (tid == 0) partials[blockIdx.x] = s_wsum[0] + s_wsum[1] + s_wsum[2] + s_wsum[3];
}

// loss = 1.25 * sum / (N*D), deterministic single-wave reduction
__global__ void vq_loss(const float* __restrict__ partials, float* __restrict__ loss_out) {
    const int l = threadIdx.x;
    float s = 0.f;
    for (int i = l; i < MAIN_BLOCKS; i += 32) s += partials[i];
    #pragma unroll
    for (int off = 16; off; off >>= 1) s += __shfl_xor(s, off, 32);
    if (l == 0) *loss_out = s * (1.25f / 16777216.f);
}

extern "C" void kernel_launch(void* const* d_in, const int* in_sizes, int n_in,
                              void* d_out, int out_size, void* d_ws, size_t ws_size,
                              hipStream_t stream) {
    (void)in_sizes; (void)n_in; (void)ws_size;
    const float* z   = (const float*)d_in[0];
    const float* emb = (const float*)d_in[1];
    float* out = (float*)d_out;

    // ws layout: enorm[1024] f32 | partials[1024] f32 | embB[262144] u16  (~520 KB)
    float* enorm    = (float*)d_ws;
    float* partials = enorm + K_EMB;
    unsigned short* embB = (unsigned short*)(partials + 1024);
    float* loss_out = out + ((size_t)out_size - 1);

    vq_enorm <<<K_EMB, 32, 0, stream>>>(emb, enorm);
    vq_pack_b<<<(K_EMB * D_DIM) / 256, 256, 0, stream>>>(emb, embB);
    vq_main  <<<MAIN_BLOCKS, 128, 0, stream>>>(z, emb, embB, enorm, out, partials);
    vq_loss  <<<1, 32, 0, stream>>>(partials, loss_out);
}
